// MultiBoxLoss_3092376453531
// MI455X (gfx1250) — compile-verified, matrix operand order checked by reading, not run on hardware
//
#include <hip/hip_runtime.h>

#define NB 128
#define NP 8732
#define NO 16
#define NC 21
#define TILE 256
#define NTILES ((NP + TILE - 1) / TILE) /* 35 */
#define TELEM (TILE * NC)               /* 5376 */

typedef __attribute__((ext_vector_type(16))) _Float16     v16h;
typedef __attribute__((ext_vector_type(8)))  float        v8f;
typedef __attribute__((ext_vector_type(4)))  unsigned int u32x4;
typedef __attribute__((ext_vector_type(8)))  int          i32x8;
typedef __attribute__((ext_vector_type(4)))  int          i32x4;

#ifndef __has_builtin
#define __has_builtin(x) 0
#endif
#if defined(__HIP_DEVICE_COMPILE__) && __has_builtin(__builtin_amdgcn_tensor_load_to_lds) && __has_builtin(__builtin_amdgcn_s_wait_tensorcnt)
#define USE_TDM 1
#else
#define USE_TDM 0
#endif

// Order-preserving float->uint key (ascending): larger float => larger key.
__device__ __forceinline__ unsigned fkey(float f) {
  unsigned u = __float_as_uint(f);
  return (u & 0x80000000u) ? ~u : (u | 0x80000000u);
}

// ---------------------------------------------------------------------------
// K1: per-row matching, conf targets, smooth-L1 loc loss partials
// ---------------------------------------------------------------------------
__global__ __launch_bounds__(256) void k_match(
    const float* __restrict__ loc_preds, const float* __restrict__ gt_data,
    const float* __restrict__ priors, int* __restrict__ conf_out,
    int* __restrict__ numpos_out, float* __restrict__ locloss_out) {
  const int b = blockIdx.x, tid = threadIdx.x;
  __shared__ float s_gt[NO * 5];
  __shared__ unsigned long long s_best[NO];
  __shared__ int s_bpi[NO];
  __shared__ float s_redf[256];
  __shared__ int s_redi[256];

  if (tid < NO * 5) s_gt[tid] = gt_data[b * NO * 5 + tid];
  if (tid < NO) s_best[tid] = 0ull;
  __syncthreads();

  // Pass A: best prior per GT (argmax over priors; ties -> lowest p via ~p).
  unsigned long long lbest[NO];
#pragma unroll
  for (int g = 0; g < NO; g++) lbest[g] = 0ull;

  for (int p = tid; p < NP; p += 256) {
    const float pcx = priors[p * 4 + 0], pcy = priors[p * 4 + 1];
    const float pw = priors[p * 4 + 2], ph = priors[p * 4 + 3];
    const float pl = pcx - 0.5f * pw, pt = pcy - 0.5f * ph;
    const float pr = pcx + 0.5f * pw, pb = pcy + 0.5f * ph;
    const float areaP = pw * ph;
#pragma unroll
    for (int g = 0; g < NO; g++) {
      float gx0 = s_gt[g * 5 + 0], gy0 = s_gt[g * 5 + 1];
      float gx1 = s_gt[g * 5 + 2], gy1 = s_gt[g * 5 + 3];
      float iw = fmaxf(fminf(pr, gx1) - fmaxf(pl, gx0), 0.f);
      float ih = fmaxf(fminf(pb, gy1) - fmaxf(pt, gy0), 0.f);
      float inter = iw * ih;
      float iou = inter / ((gx1 - gx0) * (gy1 - gy0) + areaP - inter);
      unsigned long long key =
          ((unsigned long long)__float_as_uint(iou) << 32) |
          (unsigned long long)(0xFFFFFFFFu - (unsigned)p);
      if (key > lbest[g]) lbest[g] = key;
    }
  }
#pragma unroll
  for (int g = 0; g < NO; g++) atomicMax(&s_best[g], lbest[g]);
  __syncthreads();
  if (tid < NO)
    s_bpi[tid] = (int)(0xFFFFFFFFu - (unsigned)(s_best[tid] & 0xFFFFFFFFull));
  __syncthreads();

  // Pass B: best GT per prior + override; conf + loc loss.
  float locSum = 0.f;
  int posCnt = 0;
  for (int p = tid; p < NP; p += 256) {
    const float pcx = priors[p * 4 + 0], pcy = priors[p * 4 + 1];
    const float pw = priors[p * 4 + 2], ph = priors[p * 4 + 3];
    const float pl = pcx - 0.5f * pw, pt = pcy - 0.5f * ph;
    const float pr = pcx + 0.5f * pw, pb = pcy + 0.5f * ph;
    const float areaP = pw * ph;
    float btv = -1.f;
    int bti = 0;
#pragma unroll
    for (int g = 0; g < NO; g++) {
      float gx0 = s_gt[g * 5 + 0], gy0 = s_gt[g * 5 + 1];
      float gx1 = s_gt[g * 5 + 2], gy1 = s_gt[g * 5 + 3];
      float iw = fmaxf(fminf(pr, gx1) - fmaxf(pl, gx0), 0.f);
      float ih = fmaxf(fminf(pb, gy1) - fmaxf(pt, gy0), 0.f);
      float inter = iw * ih;
      float iou = inter / ((gx1 - gx0) * (gy1 - gy0) + areaP - inter);
      if (iou > btv) { btv = iou; bti = g; }  // first-occurrence argmax
    }
#pragma unroll
    for (int g = 0; g < NO; g++)
      if (s_bpi[g] == p) { btv = 2.0f; bti = g; }  // last GT wins
    int cf = (btv < 0.5f) ? 0 : (int)s_gt[bti * 5 + 4];
    conf_out[b * NP + p] = cf;
    if (cf > 0) {
      posCnt++;
      float m0 = s_gt[bti * 5 + 0], m1 = s_gt[bti * 5 + 1];
      float m2 = s_gt[bti * 5 + 2], m3 = s_gt[bti * 5 + 3];
      float tv[4];
      tv[0] = ((m0 + m2) * 0.5f - pcx) * 10.f / pw;  // /(0.1*pw)
      tv[1] = ((m1 + m3) * 0.5f - pcy) * 10.f / ph;
      tv[2] = logf((m2 - m0) / pw) * 5.f;            // /0.2
      tv[3] = logf((m3 - m1) / ph) * 5.f;
      const float* lp = loc_preds + ((long long)b * NP + p) * 4;
#pragma unroll
      for (int i = 0; i < 4; i++) {
        float d = fabsf(lp[i] - tv[i]);
        locSum += (d < 1.f) ? 0.5f * d * d : d - 0.5f;
      }
    }
  }
  s_redf[tid] = locSum;
  s_redi[tid] = posCnt;
  __syncthreads();
  for (int st = 128; st > 0; st >>= 1) {
    if (tid < st) {
      s_redf[tid] += s_redf[tid + st];
      s_redi[tid] += s_redi[tid + st];
    }
    __syncthreads();
  }
  if (tid == 0) {
    locloss_out[b] = s_redf[0];
    numpos_out[b] = s_redi[0];
  }
}

// ---------------------------------------------------------------------------
// K2: cross-entropy via TDM staging + WMMA exp-row-sums
// ---------------------------------------------------------------------------
__global__ __launch_bounds__(256) void k_ce(
    const float* __restrict__ scores, const int* __restrict__ conf_in,
    float* __restrict__ ce_out, unsigned* __restrict__ key_out,
    float* __restrict__ posce_out) {
  const int b = blockIdx.y, tile = blockIdx.x, tid = threadIdx.x;
  const int pbase = tile * TILE;
  __shared__ float s_sc[TELEM];
  __shared__ float s_sums[8][32];
  __shared__ float s_red[256];

  const long long tileElemBase = ((long long)b * NP + pbase) * NC;
  const long long totalElems = (long long)NB * NP * NC;

#if USE_TDM
  if (tid < 32) {
    unsigned long long gaddr =
        (unsigned long long)(const void*)scores + (unsigned long long)tileElemBase * 4ull;
    unsigned ldsa = (unsigned)(unsigned long long)(void*)&s_sc[0];
    long long remll = totalElems - tileElemBase;
    unsigned rem32 = (unsigned)((remll > 0x7FFFFFFFll) ? 0x7FFFFFFFll : remll);
    u32x4 g0;
    g0[0] = 1u;                                             // count=1
    g0[1] = ldsa;                                           // lds_addr
    g0[2] = (unsigned)(gaddr & 0xFFFFFFFFull);              // global_addr lo
    g0[3] = (unsigned)((gaddr >> 32) & 0x01FFFFFFull) | (2u << 30);  // hi + type=2
    i32x8 g1;
    g1[0] = (int)(2u << 16);                                // data_size=4B, mask=0
    g1[1] = (int)((rem32 & 0xFFFFu) << 16);                 // tensor_dim0[15:0]
    g1[2] = (int)((rem32 >> 16) | (1u << 16));              // tensor_dim0 hi, dim1=1
    g1[3] = (int)(((unsigned)TELEM & 0xFFFFu) << 16);       // tile_dim0
    g1[4] = 0;                                              // tile_dim1/2 = 0 (1-D)
    g1[5] = (int)TELEM;                                     // tensor_dim0_stride
    g1[6] = 0;
    g1[7] = 0;
    i32x4 gz;
    gz[0] = 0; gz[1] = 0; gz[2] = 0; gz[3] = 0;
#if __clang_major__ >= 23
    i32x8 gz8;
    for (int i = 0; i < 8; i++) gz8[i] = 0;
    __builtin_amdgcn_tensor_load_to_lds(g0, g1, gz, gz, gz8, 0);
#else
    __builtin_amdgcn_tensor_load_to_lds(g0, g1, gz, gz, 0);
#endif
    __builtin_amdgcn_s_wait_tensorcnt(0);
  }
#else
  for (int i = tid; i < TELEM; i += 256) {
    long long gi = tileElemBase + i;
    s_sc[i] = (gi < totalElems) ? scores[gi] : 0.f;
  }
#endif
  __syncthreads();

  const int lane = tid & 31, wave = tid >> 5;
  const int hi = lane >> 4;       // K-half selector per 16-bit A-matrix layout
  const int row = lane & 15;      // M row
  float myM[2];

#pragma unroll
  for (int q = 0; q < 2; q++) {
    const int lp = wave * 32 + q * 16 + row;
    const float* sp = &s_sc[lp * NC];
    float lm = -3.0e38f;
    float vals[16];
    // Branch-free: always load a clamped in-range LDS address, then select the
    // value; avoids exec-mask save/restore around per-element ds_loads.
#pragma unroll
    for (int e = 0; e < 16; e++) {
      int K = (e < 8) ? ((hi ? 8 : 0) + e) : ((hi ? 24 : 16) + (e - 8));
      int Kc = (K < NC) ? K : (NC - 1);
      float raw = sp[Kc];
      float v = (K < NC) ? raw : -3.0e38f;
      vals[e] = v;
      lm = fmaxf(lm, v);
    }
    float pm = __shfl_xor(lm, 16);       // partner holds the other K-half
    float M = fmaxf(lm, pm);
    myM[q] = M;
    v16h a;
#pragma unroll
    for (int e = 0; e < 16; e++) {
      float ev = __builtin_expf(vals[e] - M);
      int K = (e < 8) ? ((hi ? 8 : 0) + e) : ((hi ? 24 : 16) + (e - 8));
      a[e] = (K < NC) ? (_Float16)ev : (_Float16)0.f;
    }
    v16h ones;
#pragma unroll
    for (int e = 0; e < 16; e++) ones[e] = (_Float16)1.f;
    v8f c = {};
    // D = A(16x32 f16) x ones(32x16) + 0 : every column of D = row sums of A
    c = __builtin_amdgcn_wmma_f32_16x16x32_f16(false, a, false, ones, (short)0,
                                               c, false, false);
#pragma unroll
    for (int r = 0; r < 8; r++) s_sums[wave][q * 16 + r + hi * 8] = c[r];
  }
  __syncthreads();

  const int lp = wave * 32 + lane;
  const int p = pbase + lp;
  if (p < NP) {
    float M = (lane < 16) ? myM[0] : myM[1];
    float lse = M + logf(s_sums[wave][lane]);
    int ct = conf_in[b * NP + p];
    float gathered = s_sc[lp * NC + ct];
    float ce = lse - gathered;
    ce_out[b * NP + p] = ce;
    bool pos = ct > 0;
    float mine = pos ? 0.f : ce;
    key_out[b * NP + p] = fkey(mine);
    s_red[tid] = pos ? ce : 0.f;
  } else {
    s_red[tid] = 0.f;
  }
  __syncthreads();
  for (int st = 128; st > 0; st >>= 1) {
    if (tid < st) s_red[tid] += s_red[tid + st];
    __syncthreads();
  }
  if (tid == 0) posce_out[b * NTILES + tile] = s_red[0];
}

// ---------------------------------------------------------------------------
// K3a: total positives
// ---------------------------------------------------------------------------
__global__ void k_totpos(const int* __restrict__ numpos, int* __restrict__ totpos) {
  __shared__ int s[128];
  int t = threadIdx.x;
  s[t] = numpos[t];
  __syncthreads();
  for (int st = 64; st > 0; st >>= 1) {
    if (t < st) s[t] += s[t + st];
    __syncthreads();
  }
  if (t == 0) totpos[0] = s[0];
}

// ---------------------------------------------------------------------------
// K3b: per-row top-K (radix select + stable tie pass) -> negative CE sum
// ---------------------------------------------------------------------------
__global__ __launch_bounds__(256) void k_select(
    const unsigned* __restrict__ key, const float* __restrict__ ce,
    const int* __restrict__ conf, const int* __restrict__ numpos,
    const int* __restrict__ totpos, float* __restrict__ negce_out) {
  const int b = blockIdx.x, tid = threadIdx.x;
  __shared__ unsigned s_hist[256];
  __shared__ unsigned s_scan[256];
  __shared__ unsigned s_prefix, s_remaining, s_running;
  __shared__ float s_red[256];

  long long Kll = (long long)3 * numpos[b];
  long long lim = (long long)NP - totpos[0] - 1;
  if (lim < Kll) Kll = lim;
  if (Kll <= 0) {
    if (tid == 0) negce_out[b] = 0.f;
    return;
  }
  const unsigned Ku = (unsigned)Kll;
  const unsigned* kb = key + (long long)b * NP;
  const float* cb = ce + (long long)b * NP;
  const int* fb = conf + (long long)b * NP;

  if (tid == 0) { s_prefix = 0u; s_remaining = Ku; }
  __syncthreads();

  for (int shift = 24; shift >= 0; shift -= 8) {
    s_hist[tid] = 0u;
    __syncthreads();
    unsigned pref = s_prefix;
    unsigned hmask = (shift == 24) ? 0u : (0xFFFFFFFFu << (shift + 8));
    for (int p = tid; p < NP; p += 256) {
      unsigned k = kb[p];
      if ((k & hmask) == (pref & hmask))
        atomicAdd(&s_hist[(k >> shift) & 255u], 1u);
    }
    __syncthreads();
    if (tid == 0) {
      unsigned rem = s_remaining, run = 0, sel = 0;
      for (int v = 255; v >= 0; v--) {
        unsigned c = s_hist[v];
        if (run + c >= rem) { sel = (unsigned)v; break; }
        run += c;
      }
      s_prefix = pref | (sel << shift);
      s_remaining = rem - run;
    }
    __syncthreads();
  }
  const unsigned T = s_prefix;
  const unsigned tiesToTake = s_remaining;
  if (tid == 0) s_running = 0u;
  __syncthreads();

  float partial = 0.f;
  for (int base = 0; base < NP; base += 256) {
    int p = base + tid;
    unsigned k = (p < NP) ? kb[p] : 0u;
    float cv = (p < NP) ? cb[p] : 0.f;
    bool isPos = (p < NP) && (fb[p] > 0);
    if (p < NP && k > T && !isPos) partial += cv;
    unsigned tie = (p < NP && k == T) ? 1u : 0u;
    s_scan[tid] = tie;
    __syncthreads();
    for (int off = 1; off < 256; off <<= 1) {
      unsigned v = (tid >= (unsigned)off) ? s_scan[tid - off] : 0u;
      __syncthreads();
      s_scan[tid] += v;
      __syncthreads();
    }
    unsigned incl = s_scan[tid];
    unsigned excl = incl - tie;
    unsigned runbase = s_running;
    if (tie && (runbase + excl) < tiesToTake && !isPos) partial += cv;
    __syncthreads();
    if (tid == 255) s_running = runbase + s_scan[255];
    __syncthreads();
  }
  s_red[tid] = partial;
  __syncthreads();
  for (int st = 128; st > 0; st >>= 1) {
    if (tid < st) s_red[tid] += s_red[tid + st];
    __syncthreads();
  }
  if (tid == 0) negce_out[b] = s_red[0];
}

// ---------------------------------------------------------------------------
// K4: final deterministic combine -> (loc_loss/N, conf_loss/N)
// ---------------------------------------------------------------------------
__global__ void k_final(const float* __restrict__ locloss,
                        const float* __restrict__ posce,
                        const float* __restrict__ negce,
                        const int* __restrict__ totpos, float* __restrict__ out) {
  __shared__ float s[256];
  __shared__ float res[2];
  int t = threadIdx.x;
  float v = 0.f;
  for (int i = t; i < NB; i += 256) v += locloss[i];
  s[t] = v;
  __syncthreads();
  for (int st = 128; st > 0; st >>= 1) { if (t < st) s[t] += s[t + st]; __syncthreads(); }
  if (t == 0) res[0] = s[0];
  __syncthreads();
  v = 0.f;
  for (int i = t; i < NB * NTILES; i += 256) v += posce[i];
  s[t] = v;
  __syncthreads();
  for (int st = 128; st > 0; st >>= 1) { if (t < st) s[t] += s[t + st]; __syncthreads(); }
  if (t == 0) res[1] = s[0];
  __syncthreads();
  v = 0.f;
  for (int i = t; i < NB; i += 256) v += negce[i];
  s[t] = v;
  __syncthreads();
  for (int st = 128; st > 0; st >>= 1) { if (t < st) s[t] += s[t + st]; __syncthreads(); }
  if (t == 0) {
    float N = (float)totpos[0];
    out[0] = res[0] / N;
    out[1] = (res[1] + s[0]) / N;
  }
}

// ---------------------------------------------------------------------------
extern "C" void kernel_launch(void* const* d_in, const int* in_sizes, int n_in,
                              void* d_out, int out_size, void* d_ws, size_t ws_size,
                              hipStream_t stream) {
  const float* loc_preds = (const float*)d_in[0];
  const float* score_preds = (const float*)d_in[1];
  const float* gt_data = (const float*)d_in[2];
  const float* priors = (const float*)d_in[3];
  float* out = (float*)d_out;

  int* ws_conf = (int*)d_ws;                       // NB*NP
  float* ws_ce = (float*)(ws_conf + NB * NP);      // NB*NP
  unsigned* ws_key = (unsigned*)(ws_ce + NB * NP); // NB*NP
  int* ws_numpos = (int*)(ws_key + NB * NP);       // NB
  float* ws_locloss = (float*)(ws_numpos + NB);    // NB
  float* ws_posce = ws_locloss + NB;               // NB*NTILES
  float* ws_negce = ws_posce + NB * NTILES;        // NB
  int* ws_totpos = (int*)(ws_negce + NB);          // 1

  k_match<<<NB, 256, 0, stream>>>(loc_preds, gt_data, priors, ws_conf,
                                  ws_numpos, ws_locloss);
  k_ce<<<dim3(NTILES, NB), 256, 0, stream>>>(score_preds, ws_conf, ws_ce,
                                             ws_key, ws_posce);
  k_totpos<<<1, 128, 0, stream>>>(ws_numpos, ws_totpos);
  k_select<<<NB, 256, 0, stream>>>(ws_key, ws_ce, ws_conf, ws_numpos,
                                   ws_totpos, ws_negce);
  k_final<<<1, 256, 0, stream>>>(ws_locloss, ws_posce, ws_negce, ws_totpos, out);
}
